// VectorQuantizer_61838939128180
// MI455X (gfx1250) — matrix operand order repaired and checked by reading, not compiled, not run on hardware
//
#include <hip/hip_runtime.h>
#include <hip/hip_bf16.h>

// Problem constants (from reference): B=16, D=256, H=32, W=32, K=8192
#define BATCH   16
#define DDIM    256
#define HH      32
#define WW      32
#define KCODES  8192
#define NTOK    (BATCH * HH * WW)        // 16384
#define OUTELEM (BATCH * DDIM * HH * WW) // 4194304

typedef __attribute__((ext_vector_type(16))) __bf16 v16bf;
typedef __attribute__((ext_vector_type(8)))  __bf16 v8bf;
typedef __attribute__((ext_vector_type(8)))  float  v8f;

static __device__ __forceinline__ __bf16 f2bf(float f) {
    union { float f; unsigned u; } v; v.f = f;
    unsigned r = v.u + 0x7FFFu + ((v.u >> 16) & 1u);   // round-to-nearest-even
    unsigned short h = (unsigned short)(r >> 16);
    return __builtin_bit_cast(__bf16, h);
}

// CDNA5 async global->LDS copy (ASYNCcnt-tracked, no VGPR round trip).
// GVS addressing: mem = SGPR64 + VGPR32 + ioffset; LDS dst = VGPR[vdst].
static __device__ __forceinline__ void async_b128(unsigned ldsAddr,
                                                  unsigned long long gbase,
                                                  unsigned goff) {
    asm volatile("global_load_async_to_lds_b128 %0, %1, %2"
                 :
                 : "v"(ldsAddr), "v"(goff), "s"(gbase)
                 : "memory");
}

static __device__ __forceinline__ void wait_async0() {
#if __has_builtin(__builtin_amdgcn_s_wait_asynccnt)
    __builtin_amdgcn_s_wait_asynccnt(0);
#else
    asm volatile("s_wait_asynccnt 0x0" ::: "memory");
#endif
}

// Copy n16*4KB (n16 b128 per thread, 256 threads) global -> LDS asynchronously.
static __device__ __forceinline__ void async_chunk_copy(unsigned ldsBase,
                                                        const __bf16* gsrc,
                                                        int tid, int n16) {
    unsigned long long gb = (unsigned long long)(uintptr_t)gsrc;
    #pragma unroll
    for (int i = 0; i < 16; ++i) {
        if (i >= n16) break;
        unsigned off = (unsigned)(tid * 16 + i * 4096);
        async_b128(ldsBase + off, gb, off);
    }
}

// ---------------------------------------------------------------------------
// Kernel 1: x [B][D][H][W] f32  ->  xt [N][D] bf16   (N = b*1024 + h*32 + w)
// ---------------------------------------------------------------------------
__global__ __launch_bounds__(1024) void vq_transpose_x(const float* __restrict__ x,
                                                       __bf16* __restrict__ xt) {
    __shared__ float s[32][33];
    const int b   = blockIdx.z;
    const int d0  = blockIdx.y * 32;
    const int hw0 = blockIdx.x * 32;
    const int tx = threadIdx.x, ty = threadIdx.y;
    s[ty][tx] = x[(size_t)(b * DDIM + d0 + ty) * (HH * WW) + hw0 + tx];
    __syncthreads();
    xt[(size_t)(b * (HH * WW) + hw0 + ty) * DDIM + d0 + tx] = f2bf(s[tx][ty]);
}

// ---------------------------------------------------------------------------
// Kernel 2: E [D][K] f32  ->  et [K][D] bf16
// ---------------------------------------------------------------------------
__global__ __launch_bounds__(1024) void vq_transpose_e(const float* __restrict__ e,
                                                       __bf16* __restrict__ et) {
    __shared__ float s[32][33];
    const int k0 = blockIdx.x * 32;
    const int d0 = blockIdx.y * 32;
    const int tx = threadIdx.x, ty = threadIdx.y;
    s[ty][tx] = e[(size_t)(d0 + ty) * KCODES + k0 + tx];
    __syncthreads();
    et[(size_t)(k0 + ty) * DDIM + d0 + tx] = f2bf(s[tx][ty]);
}

// ---------------------------------------------------------------------------
// Kernel 3: normE[k] = sum_d E[d][k]^2
// ---------------------------------------------------------------------------
__global__ __launch_bounds__(256) void vq_norm_e(const float* __restrict__ e,
                                                 float* __restrict__ normE) {
    const int k = blockIdx.x * 256 + threadIdx.x;
    float s = 0.0f;
    #pragma unroll 8
    for (int d = 0; d < DDIM; ++d) {
        float v = e[(size_t)d * KCODES + k];
        s += v * v;
    }
    normE[k] = s;
}

// ---------------------------------------------------------------------------
// Kernel 4: fused GEMM (bf16 WMMA, f32 accum) + running argmin over K
//   block = 256 threads (8 waves); block owns M_TILE=128 tokens.
//   K loop in chunks of 64 codes; B chunks double-buffered in LDS and staged
//   with global_load_async_to_lds_b128 (overlaps copy of chunk k+1 with the
//   WMMAs of chunk k). LDS: A 64KB + 2x B 32KB = 128KB -> 2 WGs per WGP.
// ---------------------------------------------------------------------------
#define M_TILE 128
#define N_CHUNK 64
#define N_CHUNKS (KCODES / N_CHUNK)          // 128
#define NT_TILES (N_CHUNK / 16)              // 4
#define A_HALVES (M_TILE * DDIM)             // 32768 halves = 64KB
#define B_HALVES (N_CHUNK * DDIM)            // 16384 halves = 32KB
#define LDS_B0   ((unsigned)(A_HALVES * 2))          // 65536
#define LDS_B1   ((unsigned)(A_HALVES * 2 + B_HALVES * 2)) // 98304

__global__ __launch_bounds__(256) void vq_argmin(const __bf16* __restrict__ xt,
                                                 const __bf16* __restrict__ et,
                                                 const float* __restrict__ normE,
                                                 int* __restrict__ indOut) {
    extern __shared__ __bf16 smem[];
    const __bf16* As = smem;                         // [128][256]
    const __bf16* B0 = smem + A_HALVES;              // [64][256]
    const __bf16* B1 = smem + A_HALVES + B_HALVES;   // [64][256]

    const int tid  = threadIdx.x;
    const int wave = tid >> 5;
    const int lane = tid & 31;
    const int l16  = lane & 15;
    const int hi   = (lane >> 4) & 1;       // 0: lanes 0-15, 1: lanes 16-31

    // Kick off async staging: A tile (64KB) + first B chunk (32KB).
    async_chunk_copy(0u, xt + (size_t)blockIdx.x * M_TILE * DDIM, tid, 16);
    async_chunk_copy(LDS_B0, et, tid, 8);

    float minv[8];
    int   mini[8];
    #pragma unroll
    for (int j = 0; j < 8; ++j) { minv[j] = 3.4e38f; mini[j] = 0; }

    const __bf16* aRow = As + (size_t)(wave * 16 + l16) * DDIM + hi * 8;

    for (int kc = 0; kc < N_CHUNKS; ++kc) {
        wait_async0();       // this wave's async copies landed in LDS
        __syncthreads();     // everyone's copies visible; prev compute done

        // Overlap: stage chunk kc+1 into the other buffer while computing kc.
        if (kc + 1 < N_CHUNKS) {
            async_chunk_copy((kc & 1) ? LDS_B0 : LDS_B1,
                             et + (size_t)(kc + 1) * B_HALVES, tid, 8);
        }
        const __bf16* Bb = (kc & 1) ? B1 : B0;
        const __bf16* bRow = Bb + (size_t)l16 * DDIM + hi * 8;

        v8f acc[NT_TILES];
        const v8f zero = {0.f, 0.f, 0.f, 0.f, 0.f, 0.f, 0.f, 0.f};
        #pragma unroll
        for (int nt = 0; nt < NT_TILES; ++nt) acc[nt] = zero;

        // Software-pipelined fragment loads: fetch kt+1 while computing kt.
        v16bf a_cur, a_nxt;
        v16bf b_cur[NT_TILES], b_nxt[NT_TILES];
        {
            v8bf alo = *(const v8bf*)(aRow);
            v8bf ahi = *(const v8bf*)(aRow + 16);
            a_cur = __builtin_shufflevector(alo, ahi,
                0,1,2,3,4,5,6,7,8,9,10,11,12,13,14,15);
            #pragma unroll
            for (int nt = 0; nt < NT_TILES; ++nt) {
                const __bf16* bp = bRow + nt * 16 * DDIM;
                v8bf b0v = *(const v8bf*)(bp);
                v8bf b1v = *(const v8bf*)(bp + 16);
                b_cur[nt] = __builtin_shufflevector(b0v, b1v,
                    0,1,2,3,4,5,6,7,8,9,10,11,12,13,14,15);
            }
        }
        #pragma unroll
        for (int kt = 0; kt < 8; ++kt) {
            if (kt < 7) {
                const int kb = (kt + 1) * 32;
                v8bf alo = *(const v8bf*)(aRow + kb);
                v8bf ahi = *(const v8bf*)(aRow + kb + 16);
                a_nxt = __builtin_shufflevector(alo, ahi,
                    0,1,2,3,4,5,6,7,8,9,10,11,12,13,14,15);
                #pragma unroll
                for (int nt = 0; nt < NT_TILES; ++nt) {
                    const __bf16* bp = bRow + nt * 16 * DDIM + kb;
                    v8bf b0v = *(const v8bf*)(bp);
                    v8bf b1v = *(const v8bf*)(bp + 16);
                    b_nxt[nt] = __builtin_shufflevector(b0v, b1v,
                        0,1,2,3,4,5,6,7,8,9,10,11,12,13,14,15);
                }
            }
            #pragma unroll
            for (int nt = 0; nt < NT_TILES; ++nt) {
                acc[nt] = __builtin_amdgcn_wmma_f32_16x16x32_bf16(
                    false, a_cur, false, b_cur[nt], (short)0, acc[nt],
                    false, false);
            }
            a_cur = a_nxt;
            #pragma unroll
            for (int nt = 0; nt < NT_TILES; ++nt) b_cur[nt] = b_nxt[nt];
        }

        // score = ||E_k||^2 - 2 <x, E_k>; per-lane codes strictly increase,
        // so strict '<' keeps the first (lowest-index) minimum.
        #pragma unroll
        for (int nt = 0; nt < NT_TILES; ++nt) {
            const int code = kc * N_CHUNK + nt * 16 + l16;
            const float nE = normE[code];
            #pragma unroll
            for (int j = 0; j < 8; ++j) {
                float sc = __builtin_fmaf(-2.0f, acc[nt][j], nE);
                if (sc < minv[j]) { minv[j] = sc; mini[j] = code; }
            }
        }
    }

    // cross-lane argmin: C/D layout puts token M=j(+8) on VGPR j, code on lane;
    // reduce over the 16 lanes within each half-wave group.
    #pragma unroll
    for (int s = 1; s < 16; s <<= 1) {
        #pragma unroll
        for (int j = 0; j < 8; ++j) {
            float ov = __shfl_xor(minv[j], s, 32);
            int   oi = __shfl_xor(mini[j], s, 32);
            if (ov < minv[j] || (ov == minv[j] && oi < mini[j])) {
                minv[j] = ov; mini[j] = oi;
            }
        }
    }

    if (l16 == 0) {
        const int base = blockIdx.x * M_TILE + wave * 16 + hi * 8;
        #pragma unroll
        for (int j = 0; j < 8; ++j) indOut[base + j] = mini[j];
    }
}

// ---------------------------------------------------------------------------
// Kernel 5: gather codebook columns into both outputs + emit indices
//   out[0 .. OUTELEM)        = quantized_x_d (== quantized_x numerically)
//   out[OUTELEM .. 2*OUTELEM)= quantized_x
//   out[2*OUTELEM .. +NTOK)  = indices (as float)
// ---------------------------------------------------------------------------
__global__ __launch_bounds__(256) void vq_gather(const float* __restrict__ e,
                                                 const int* __restrict__ ind,
                                                 float* __restrict__ out) {
    const int t = blockIdx.x * 256 + threadIdx.x;   // [b][d][h][w] linear
    const int w = t & 31;
    const int h = (t >> 5) & 31;
    const int d = (t >> 10) & 255;
    const int b = t >> 18;
    const int n = (b << 10) + (h << 5) + w;
    const int idx = ind[n];
    const float v = e[(size_t)d * KCODES + idx];
    out[t] = v;
    out[t + OUTELEM] = v;
    if (d == 0) out[2 * OUTELEM + n] = (float)idx;
}

// ---------------------------------------------------------------------------
extern "C" void kernel_launch(void* const* d_in, const int* in_sizes, int n_in,
                              void* d_out, int out_size, void* d_ws, size_t ws_size,
                              hipStream_t stream) {
    (void)in_sizes; (void)n_in; (void)out_size; (void)ws_size;
    const float* x = (const float*)d_in[0];   // [16,256,32,32]
    const float* e = (const float*)d_in[1];   // [256,8192]
    float* out = (float*)d_out;

    char* ws = (char*)d_ws;
    __bf16* xt    = (__bf16*)(ws);                          //  8 MB  [N][D] bf16
    __bf16* et    = (__bf16*)(ws + 8388608);                //  4 MB  [K][D] bf16
    float*  normE = (float*)(ws + 8388608 + 4194304);       // 32 KB  [K] f32
    int*    ind   = (int*)(ws + 8388608 + 4194304 + 32768); // 64 KB  [N] i32

    dim3 tb(32, 32);
    vq_transpose_x<<<dim3(HH * WW / 32, DDIM / 32, BATCH), tb, 0, stream>>>(x, xt);
    vq_transpose_e<<<dim3(KCODES / 32, DDIM / 32), tb, 0, stream>>>(e, et);
    vq_norm_e<<<KCODES / 256, 256, 0, stream>>>(e, normE);

    const size_t ldsBytes = (size_t)(A_HALVES + 2 * B_HALVES) * sizeof(__bf16); // 128 KB
    vq_argmin<<<NTOK / M_TILE, 256, ldsBytes, stream>>>(xt, et, normE, ind);

    vq_gather<<<OUTELEM / 256, 256, 0, stream>>>(e, ind, out);
}
